// gcn_operation_90323162234921
// MI455X (gfx1250) — matrix operand order repaired and checked by reading, not compiled, not run on hardware
//
#include <hip/hip_runtime.h>

// ---------------------------------------------------------------------------
// GCN layer, fused, for MI455X (gfx1250, wave32, WMMA bf16 16x16x32).
//
//   h = blocktridiag(adj, wct) @ x      (3072 x 2048, block-sparse K)
//   out = (h@W + b)[:, :64] * sigmoid((h@W + b)[:, 64:])
//
// Pipeline:
//   pass 1: f32 -> bf16 packing into d_ws  (adj, wct, wct^T, x^T; A-blocks
//           stored contiguously so the GEMM uses ONE base pointer -> global_load)
//   pass 2: fused WMMA GEMM (async global->LDS staging, double buffered)
//           + epilogue WMMA (h@W + bias) + GLU
// ---------------------------------------------------------------------------

typedef __attribute__((ext_vector_type(16))) __bf16 v16bf;
typedef __attribute__((ext_vector_type(8)))  float  v8f;
typedef int gcn_b128 __attribute__((vector_size(16)));   // 4 x i32, matches builtin

struct Q2     { uint4 a, b; };            // 32 bytes
struct U16x16 { unsigned short u[16]; };  // 32 bytes

#if defined(__gfx1250__) && __has_builtin(__builtin_amdgcn_global_load_async_to_lds_b128)
#define GCN_ASYNC_LDS 1
#else
#define GCN_ASYNC_LDS 0
#endif

#define GCN_GPTR(p) ((__attribute__((address_space(1))) gcn_b128*)(p))
#define GCN_LPTR(p) ((__attribute__((address_space(3))) gcn_b128*)(p))

static __device__ __forceinline__ unsigned short f32_to_bf16(float f) {
  unsigned u = __builtin_bit_cast(unsigned, f);
  unsigned r = 0x7FFFu + ((u >> 16) & 1u);          // round to nearest even
  return (unsigned short)((u + r) >> 16);
}

static __device__ __forceinline__ v16bf frag_from_q(uint4 a, uint4 b) {
  Q2 q{a, b};
  return __builtin_bit_cast(v16bf, q);
}

static __device__ __forceinline__ void gcn_wait_async0() {
#if defined(__HIP_DEVICE_COMPILE__)
# if __has_builtin(__builtin_amdgcn_s_wait_asynccnt)
  __builtin_amdgcn_s_wait_asynccnt(0);
# else
  __asm__ __volatile__("s_wait_asynccnt 0x0" ::: "memory");
# endif
#endif
}

// ------------------------- pass 1: packing kernels -------------------------

__global__ __launch_bounds__(256) void convert_copy_bf16(
    const float* __restrict__ src, unsigned short* __restrict__ dst, int n4) {
  int i = blockIdx.x * blockDim.x + threadIdx.x;
  if (i < n4) {
    float4 f = ((const float4*)src)[i];
    unsigned lo = (unsigned)f32_to_bf16(f.x) | ((unsigned)f32_to_bf16(f.y) << 16);
    unsigned hi = (unsigned)f32_to_bf16(f.z) | ((unsigned)f32_to_bf16(f.w) << 16);
    uint2 p; p.x = lo; p.y = hi;
    ((uint2*)dst)[i] = p;
  }
}

// dst[c*R + r] = bf16(src[r*C + c]); batched over blockIdx.z
__global__ __launch_bounds__(256) void transpose_convert_bf16(
    const float* __restrict__ src, unsigned short* __restrict__ dst, int R, int C) {
  __shared__ float tile[32][33];
  size_t bofs = (size_t)blockIdx.z * R * C;
  src += bofs; dst += bofs;
  int c0 = blockIdx.x * 32, r0 = blockIdx.y * 32;
  int tx = threadIdx.x, ty = threadIdx.y;
  for (int i = 0; i < 32; i += 8)
    tile[ty + i][tx] = src[(size_t)(r0 + ty + i) * C + (c0 + tx)];
  __syncthreads();
  for (int i = 0; i < 32; i += 8)
    dst[(size_t)(c0 + ty + i) * R + (r0 + tx)] = f32_to_bf16(tile[tx][ty + i]);
}

// ------------------------- pass 2: fused WMMA GEMM -------------------------
//
// Grid: (16 col-tiles of 128, 24 row-tiles of 128). 256 threads = 8 waves.
// Each wave: 32 rows x 64 cols = 8 accumulators of 16x16 f32.
// LDS (59,904 B): As/Xs double buffered [2][128][32+8] bf16, Wt[128][64+8] bf16,
// bias[128] f32.  As/Xs region reused as per-wave f32 h-buffer in the epilogue.

#define WMMA_BF16(A, B, C) \
  __builtin_amdgcn_wmma_f32_16x16x32_bf16(false, (A), false, (B), (short)0, (C), false, false)

__global__ __launch_bounds__(256) void gcn_fused_wmma(
    const unsigned short* __restrict__ xT,   // [2048][3072] bf16  (x transposed)
    const unsigned short* __restrict__ AB,   // 5 contiguous [1024][1024] bf16 blocks:
                                             //   0=adj, 1=wct0, 2=wct1, 3=wct0^T, 4=wct1^T
    const float* __restrict__ W,             // [64][128] f32
    const float* __restrict__ bias,          // [128] f32
    float* __restrict__ out)                 // [3072][32][64] f32
{
  __shared__ __align__(16) unsigned char smem[40960 + 18432 + 512];
  unsigned short* As    = (unsigned short*)smem;              // [2][128*40]
  unsigned short* Xs    = (unsigned short*)(smem + 20480);    // [2][128*40]
  unsigned short* Wt    = (unsigned short*)(smem + 40960);    // [128][72] = W^T bf16
  float*          biasS = (float*)(smem + 40960 + 18432);

  const int tid    = threadIdx.x;
  const int lane   = tid & 31, wid = tid >> 5;
  const int wave_r = wid & 3,  wave_c = wid >> 2;   // 4x2 wave grid
  const int rt = blockIdx.y, ct = blockIdx.x;
  const int blk = rt >> 3;                          // time-block 0..2
  const int rowInBlk = (rt & 7) * 128;
  const int colBase  = ct * 128;
  const int lrow = lane & 15, lhi = lane >> 4;

  // Stage W^T (bf16) and bias once.
  {
    int j = tid >> 1, half = tid & 1;
    for (int c = 0; c < 32; ++c)
      Wt[j * 72 + half * 32 + c] = f32_to_bf16(W[(size_t)(half * 32 + c) * 128 + j]);
    if (tid < 128) biasS[tid] = bias[tid];
  }

  // Block-tridiagonal K segments (A-block index within AB, x column-block offset).
  int segABlk[3], segX[3], nseg;
  if (blk == 0)      { segABlk[0] = 0; segX[0] = 0;    segABlk[1] = 1; segX[1] = 1024; nseg = 2; }
  else if (blk == 1) { segABlk[0] = 3; segX[0] = 0;    segABlk[1] = 0; segX[1] = 1024;
                       segABlk[2] = 2; segX[2] = 2048; nseg = 3; }
  else               { segABlk[0] = 4; segX[0] = 1024; segABlk[1] = 0; segX[1] = 2048; nseg = 2; }
  const int nSlab = nseg << 5;                      // 32 BK=32 slabs per segment

  const int srow = tid >> 1;                        // staging row/col 0..127
  const int soff = (tid & 1) * 16;                  // 16 bf16 = 32B per thread-half

  v8f acc[8] = {};                                  // [rs*4+cs] 16x16 f32 tiles

  auto slabPtrs = [&](int s, const unsigned short*& ga, const unsigned short*& gx) {
    int seg = s >> 5, k0 = (s & 31) << 5;
    ga = AB + ((size_t)segABlk[seg] << 20) + (size_t)(rowInBlk + srow) * 1024 + k0 + soff;
    gx = xT + (size_t)(colBase + srow) * 3072 + segX[seg] + k0 + soff;
  };

  auto computeSlab = [&](int cbuf) {
    const unsigned short* AsB = As + cbuf * 5120;
    const unsigned short* XsB = Xs + cbuf * 5120;
    v16bf af[2], bfv[4];
    for (int rs = 0; rs < 2; ++rs) {                // A frag: lane=row, one 16-K run
      const unsigned short* ap = AsB + (wave_r * 32 + rs * 16 + lrow) * 40;
      af[rs] = frag_from_q(*(const uint4*)(ap + lhi * 16),
                           *(const uint4*)(ap + lhi * 16 + 8));
    }
    for (int cs = 0; cs < 4; ++cs) {                // B frag: lane=col, one 16-K run
      const unsigned short* bp = XsB + (wave_c * 64 + cs * 16 + lrow) * 40 + lhi * 16;
      bfv[cs] = frag_from_q(((const uint4*)bp)[0], ((const uint4*)bp)[1]);
    }
    for (int rs = 0; rs < 2; ++rs)
      for (int cs = 0; cs < 4; ++cs)
        acc[rs * 4 + cs] = WMMA_BF16(af[rs], bfv[cs], acc[rs * 4 + cs]);
  };

#if GCN_ASYNC_LDS
  // ---- async global->LDS staging: DMA of slab s+1 overlaps WMMA of slab s ----
  auto issueAsync = [&](int s, int buf) {
    const unsigned short *ga, *gx;
    slabPtrs(s, ga, gx);
    unsigned short* la = As + buf * 5120 + srow * 40 + soff;
    unsigned short* lx = Xs + buf * 5120 + srow * 40 + soff;
    // Immediate offset applies to BOTH global and LDS addresses (ISA 10.x async).
    __builtin_amdgcn_global_load_async_to_lds_b128(GCN_GPTR(ga), GCN_LPTR(la), 0, 0);
    __builtin_amdgcn_global_load_async_to_lds_b128(GCN_GPTR(ga), GCN_LPTR(la), 16, 0);
    __builtin_amdgcn_global_load_async_to_lds_b128(GCN_GPTR(gx), GCN_LPTR(lx), 0, 0);
    __builtin_amdgcn_global_load_async_to_lds_b128(GCN_GPTR(gx), GCN_LPTR(lx), 16, 0);
  };

  issueAsync(0, 0);
  for (int s = 0; s < nSlab; ++s) {
    gcn_wait_async0();                 // this wave's slab-s DMA landed in LDS
    __syncthreads();                   // all waves' DMA landed; buf[(s+1)&1] free
    if (s + 1 < nSlab) issueAsync(s + 1, (s + 1) & 1);
    computeSlab(s & 1);                // wmmas consume before next barrier
  }
#else
  // ---- fallback: register-staged double buffer ----
  uint4 ra0, ra1, rx0, rx1;
  auto loadSlab = [&](int s) {
    const unsigned short *ga, *gx;
    slabPtrs(s, ga, gx);
    ra0 = ((const uint4*)ga)[0]; ra1 = ((const uint4*)ga)[1];
    rx0 = ((const uint4*)gx)[0]; rx1 = ((const uint4*)gx)[1];
  };
  auto storeSlab = [&](int buf) {
    unsigned short* la = As + buf * 5120 + srow * 40 + soff;
    unsigned short* lx = Xs + buf * 5120 + srow * 40 + soff;
    ((uint4*)la)[0] = ra0; ((uint4*)la)[1] = ra1;
    ((uint4*)lx)[0] = rx0; ((uint4*)lx)[1] = rx1;
  };
  loadSlab(0); storeSlab(0);
  __syncthreads();
  for (int s = 0; s < nSlab; ++s) {
    if (s + 1 < nSlab) loadSlab(s + 1);
    computeSlab(s & 1);
    __syncthreads();
    if (s + 1 < nSlab) storeSlab((s + 1) & 1);
    __syncthreads();
  }
#endif

  // ---------------- fused epilogue: (h @ W + b) -> GLU ----------------
  __syncthreads();                                  // done with As/Xs; reuse as hbuf
  float* hb = (float*)smem + wid * (16 * 68);       // per-wave 16x(64+4) f32
  const int b_g   = ct * 2 + wave_c;                // batch index
  const int nBase = blk * 1024 + rowInBlk + wave_r * 32;

  for (int rs = 0; rs < 2; ++rs) {
    // Spill this wave's 16x64 h sub-tile (C layout: row = lhi*8+v, col = lane%16).
    for (int cs = 0; cs < 4; ++cs) {
      v8f a = acc[rs * 4 + cs];
      int col = cs * 16 + lrow;
      for (int v = 0; v < 8; ++v)
        hb[(lhi * 8 + v) * 68 + col] = a[v];
    }
    // Same-wave LDS ops are in order: rebuild as bf16 A-fragments (K = Cin = 64).
    v16bf ha[2];
    for (int kki = 0; kki < 2; ++kki) {
      int kk = kki * 32;
      const float* hp = hb + lrow * 68;
      int c1 = kk + lhi * 16, c2 = kk + lhi * 16 + 8;
      U16x16 t;
      for (int i = 0; i < 8; ++i) t.u[i]     = f32_to_bf16(hp[c1 + i]);
      for (int i = 0; i < 8; ++i) t.u[8 + i] = f32_to_bf16(hp[c2 + i]);
      ha[kki] = __builtin_bit_cast(v16bf, t);
    }
    for (int jt = 0; jt < 4; ++jt) {
      int jl = jt * 16;
      v8f cl, cr;
      float bl = biasS[jl + lrow], br = biasS[64 + jl + lrow];
      for (int v = 0; v < 8; ++v) { cl[v] = bl; cr[v] = br; }
      for (int kki = 0; kki < 2; ++kki) {
        const unsigned short* wpl = Wt + (jl + lrow) * 72      + kki * 32 + lhi * 16;
        const unsigned short* wpr = Wt + (64 + jl + lrow) * 72 + kki * 32 + lhi * 16;
        v16bf wl = frag_from_q(((const uint4*)wpl)[0], ((const uint4*)wpl)[1]);
        v16bf wr = frag_from_q(((const uint4*)wpr)[0], ((const uint4*)wpr)[1]);
        cl = WMMA_BF16(ha[kki], wl, cl);
        cr = WMMA_BF16(ha[kki], wr, cr);
      }
      for (int v = 0; v < 8; ++v) {                 // GLU: lhs * sigmoid(rhs)
        float o = cl[v] * (1.0f / (1.0f + __expf(-cr[v])));
        int n = nBase + rs * 16 + lhi * 8 + v;
        int j = jl + lrow;
        out[((size_t)n * 32 + b_g) * 64 + j] = o;
      }
    }
  }
}

// ------------------------------ host launcher ------------------------------

extern "C" void kernel_launch(void* const* d_in, const int* in_sizes, int n_in,
                              void* d_out, int out_size, void* d_ws, size_t ws_size,
                              hipStream_t stream) {
  (void)in_sizes; (void)n_in; (void)out_size; (void)ws_size;

  const float* x   = (const float*)d_in[0];  // (3072, 32, 64)
  const float* adj = (const float*)d_in[1];  // (1024, 1024)
  const float* wct = (const float*)d_in[2];  // (2, 1024, 1024)
  const float* W   = (const float*)d_in[3];  // (64, 128)
  const float* b   = (const float*)d_in[4];  // (128,)
  float* out = (float*)d_out;

  // Workspace layout (bf16 = ushort), ~23.1 MB total.  The five A-blocks are
  // contiguous so the GEMM kernel needs only one base pointer (-> global_load).
  unsigned short* xT   = (unsigned short*)d_ws;       // [2048][3072]
  unsigned short* AB   = xT + (size_t)2048 * 3072;    // block 0: adj
  unsigned short* w0B  = AB   + (size_t)1024 * 1024;  // block 1: wct0
  unsigned short* w0tB = w0B  + (size_t)2 * 1024 * 1024;  // blocks 3,4: wct0^T, wct1^T

  convert_copy_bf16<<<1024, 256, 0, stream>>>(adj, AB, 1024 * 1024 / 4);
  convert_copy_bf16<<<2048, 256, 0, stream>>>(wct, w0B, 2 * 1024 * 1024 / 4); // blocks 1,2
  transpose_convert_bf16<<<dim3(32, 32, 2), dim3(32, 8), 0, stream>>>(wct, w0tB, 1024, 1024);
  transpose_convert_bf16<<<dim3(64, 96, 1), dim3(32, 8), 0, stream>>>(x, xT, 3072, 2048);

  gcn_fused_wmma<<<dim3(16, 24), 256, 0, stream>>>(xT, AB, W, b, out);
}
